// EncoderDecoder_30657476559097
// MI455X (gfx1250) — matrix operand order repaired and checked by reading, not compiled, not run on hardware
//
#include <hip/hip_runtime.h>

typedef __attribute__((ext_vector_type(16))) __bf16          v16bf;
typedef __attribute__((ext_vector_type(16))) unsigned short  v16u;
typedef __attribute__((ext_vector_type(8)))  float           v8f;

#define DDIM 64

static __device__ __forceinline__ unsigned short f2bf_rne(float x) {
    unsigned int u = __builtin_bit_cast(unsigned int, x);
    unsigned int r = u + 0x7FFFu + ((u >> 16) & 1u);
    return (unsigned short)(r >> 16);
}
static __device__ __forceinline__ float bf2f(unsigned short h) {
    return __builtin_bit_cast(float, (unsigned int)h << 16);
}

// ---------------------------------------------------------------------------
// Kernel 1: split centroids into bf16 hi/lo and compute ||c||^2 in f32.
// One wave32 per centroid row (D=64 -> 2 elements per lane).
// ---------------------------------------------------------------------------
__global__ __launch_bounds__(256) void vq_prep_centroids(
    const float* __restrict__ cent,
    unsigned short* __restrict__ chi,
    unsigned short* __restrict__ clo,
    float* __restrict__ c2, int K)
{
    int gtid = blockIdx.x * blockDim.x + threadIdx.x;
    int row  = gtid >> 5;
    int lane = threadIdx.x & 31;
    if (row >= K) return;

    const float* r = cent + (size_t)row * DDIM;
    float s = 0.0f;
#pragma unroll
    for (int j = 0; j < 2; ++j) {
        int d = lane + j * 32;
        float x = r[d];
        unsigned short h = f2bf_rne(x);
        float rem = x - bf2f(h);
        unsigned short l = f2bf_rne(rem);
        chi[(size_t)row * DDIM + d] = h;
        clo[(size_t)row * DDIM + d] = l;
        s = fmaf(x, x, s);
    }
#pragma unroll
    for (int m = 16; m >= 1; m >>= 1) s += __shfl_xor(s, m, 32);
    if (lane == 0) c2[row] = s;
}

// Load one 16-row A tile (rows [rowbase, rowbase+16)) into WMMA A-layout
// registers, split into bf16 hi/lo.
static __device__ __forceinline__ void load_a_tile(
    const float* __restrict__ action, int rowbase, int n, int hl,
    v16bf a_hi[2], v16bf a_lo[2])
{
    const float* arow = action + (size_t)(rowbase + n) * DDIM;
#pragma unroll
    for (int s = 0; s < 2; ++s) {
        v16u uh, ul;
#pragma unroll
        for (int run = 0; run < 2; ++run) {
            int kbase = s * 32 + hl * 16 + run * 8;
            const float4* p = (const float4*)(arow + kbase);
            float4 x0 = p[0], x1 = p[1];
            float xs[8] = {x0.x, x0.y, x0.z, x0.w, x1.x, x1.y, x1.z, x1.w};
#pragma unroll
            for (int e = 0; e < 8; ++e) {
                unsigned short h = f2bf_rne(xs[e]);
                unsigned short l = f2bf_rne(xs[e] - bf2f(h));
                uh[run * 8 + e] = h;
                ul[run * 8 + e] = l;
            }
        }
        a_hi[s] = __builtin_bit_cast(v16bf, uh);
        a_lo[s] = __builtin_bit_cast(v16bf, ul);
    }
}

// bf16x3 accumulation of one 16x16 score tile: acc += a.c (hi*hi + hi*lo + lo*hi)
static __device__ __forceinline__ v8f wmma6(
    const v16bf a_hi[2], const v16bf a_lo[2],
    v16bf b_hi0, v16bf b_hi1, v16bf b_lo0, v16bf b_lo1)
{
    v8f acc = {};
    acc = __builtin_amdgcn_wmma_f32_16x16x32_bf16(false, a_hi[0], false, b_hi0, (short)0, acc, false, false);
    acc = __builtin_amdgcn_wmma_f32_16x16x32_bf16(false, a_hi[1], false, b_hi1, (short)0, acc, false, false);
    acc = __builtin_amdgcn_wmma_f32_16x16x32_bf16(false, a_hi[0], false, b_lo0, (short)0, acc, false, false);
    acc = __builtin_amdgcn_wmma_f32_16x16x32_bf16(false, a_hi[1], false, b_lo1, (short)0, acc, false, false);
    acc = __builtin_amdgcn_wmma_f32_16x16x32_bf16(false, a_lo[0], false, b_hi0, (short)0, acc, false, false);
    acc = __builtin_amdgcn_wmma_f32_16x16x32_bf16(false, a_lo[1], false, b_hi1, (short)0, acc, false, false);
    return acc;
}

// Butterfly-merge + bins/residual emit for one 16-row tile.
static __device__ __forceinline__ void emit_tile(
    float mval[8], int midx[8], int rowbase, int lane,
    const float* __restrict__ action, const float* __restrict__ cent,
    long long* __restrict__ bins, float* __restrict__ resid)
{
#pragma unroll
    for (int x = 1; x <= 8; x <<= 1) {
#pragma unroll
        for (int r = 0; r < 8; ++r) {
            float ov = __shfl_xor(mval[r], x, 32);
            int   oi = __shfl_xor(midx[r], x, 32);
            if (ov < mval[r] || (ov == mval[r] && oi < midx[r])) {
                mval[r] = ov; midx[r] = oi;
            }
        }
    }
    // Lanes 0 / 16 hold winners for rows hl*8 + r.
#pragma unroll
    for (int r = 0; r < 8; ++r) {
        int b0 = __shfl(midx[r], 0, 32);    // row rowbase + r
        int b1 = __shfl(midx[r], 16, 32);   // row rowbase + 8 + r
        if (lane == 0) {
            bins[rowbase + r]     = (long long)b0;
            bins[rowbase + 8 + r] = (long long)b1;
        }
        int d = lane * 2;
        {
            float2 av = *(const float2*)(action + (size_t)(rowbase + r) * DDIM + d);
            float2 cv = *(const float2*)(cent + (size_t)b0 * DDIM + d);
            float2 rv; rv.x = av.x - cv.x; rv.y = av.y - cv.y;
            *(float2*)(resid + (size_t)(rowbase + r) * DDIM + d) = rv;
        }
        {
            float2 av = *(const float2*)(action + (size_t)(rowbase + 8 + r) * DDIM + d);
            float2 cv = *(const float2*)(cent + (size_t)b1 * DDIM + d);
            float2 rv; rv.x = av.x - cv.x; rv.y = av.y - cv.y;
            *(float2*)(resid + (size_t)(rowbase + 8 + r) * DDIM + d) = rv;
        }
    }
}

// ---------------------------------------------------------------------------
// Kernel 2: per-wave 32-row tile (two 16-row WMMA tiles sharing B loads).
// Sweep K centroids with bf16x3 WMMA, running argmin of (||c||^2 - 2 a.c),
// then residual + int64 bin output.
// KK > 0: compile-time centroid count (no remainder loop); KK == 0: runtime.
// ---------------------------------------------------------------------------
template <int KK>
__global__ __launch_bounds__(256) void vq_encode(
    const float* __restrict__ action,
    const unsigned short* __restrict__ chi,
    const unsigned short* __restrict__ clo,
    const float* __restrict__ c2,
    const float* __restrict__ cent,
    long long* __restrict__ bins,
    float* __restrict__ resid, int Krt)
{
    const int K        = (KK > 0) ? KK : Krt;
    const int lane     = threadIdx.x & 31;
    const int wave     = threadIdx.x >> 5;
    const int tile     = blockIdx.x * 8 + wave;
    const int rowbase0 = tile * 32;
    const int rowbase1 = rowbase0 + 16;
    const int n        = lane & 15;   // B column / A row within tile
    const int hl       = lane >> 4;   // lane half per ISA register layouts

    // A tiles resident in VGPRs for the whole K sweep.
    v16bf a0_hi[2], a0_lo[2], a1_hi[2], a1_lo[2];
    load_a_tile(action, rowbase0, n, hl, a0_hi, a0_lo);
    load_a_tile(action, rowbase1, n, hl, a1_hi, a1_lo);

    // Per-lane running argmin over centroid residue class k === n (mod 16);
    // k visited in increasing order -> strict '<' = first-occurrence argmin.
    float mval0[8], mval1[8];
    int   midx0[8], midx1[8];
#pragma unroll
    for (int r = 0; r < 8; ++r) {
        mval0[r] = 3.4e38f; midx0[r] = 0;
        mval1[r] = 3.4e38f; midx1[r] = 0;
    }

    // Incrementing pointer streams (strength-reduced addressing).
    // B in this code's contiguous k order (same as A): lane holds K run [hl*16, hl*16+16) of column n.
    const unsigned short* bh = chi + (size_t)n * DDIM + hl * 16;
    const unsigned short* bl = clo + (size_t)n * DDIM + hl * 16;
    const float*          cp = c2 + n;

#pragma unroll 2
    for (int k0 = 0; k0 < K; k0 += 16) {
        v16bf b_hi0 = *(const v16bf*)(bh);
        v16bf b_hi1 = *(const v16bf*)(bh + 32);
        v16bf b_lo0 = *(const v16bf*)(bl);
        v16bf b_lo1 = *(const v16bf*)(bl + 32);
        float cc = *cp;

        v8f acc0 = wmma6(a0_hi, a0_lo, b_hi0, b_hi1, b_lo0, b_lo1);
        v8f acc1 = wmma6(a1_hi, a1_lo, b_hi0, b_hi1, b_lo0, b_lo1);

        int col = k0 + n;
#pragma unroll
        for (int r = 0; r < 8; ++r) {
            float s0 = fmaf(-2.0f, acc0[r], cc);   // ||c||^2 - 2 a.c
            if (s0 < mval0[r]) { mval0[r] = s0; midx0[r] = col; }
            float s1 = fmaf(-2.0f, acc1[r], cc);
            if (s1 < mval1[r]) { mval1[r] = s1; midx1[r] = col; }
        }

        bh += 16 * DDIM;
        bl += 16 * DDIM;
        cp += 16;
    }

    emit_tile(mval0, midx0, rowbase0, lane, action, cent, bins, resid);
    emit_tile(mval1, midx1, rowbase1, lane, action, cent, bins, resid);
}

// ---------------------------------------------------------------------------
extern "C" void kernel_launch(void* const* d_in, const int* in_sizes, int n_in,
                              void* d_out, int out_size, void* d_ws, size_t ws_size,
                              hipStream_t stream)
{
    const float* action = (const float*)d_in[0];
    const float* cent   = (const float*)d_in[1];
    const int D = DDIM;
    const int N = in_sizes[0] / D;   // 65536 rows
    const int K = in_sizes[1] / D;   // 2048 centroids

    // Workspace layout: chi[K*D] bf16 | clo[K*D] bf16 | c2[K] f32  (~520 KB)
    unsigned short* chi = (unsigned short*)d_ws;
    unsigned short* clo = chi + (size_t)K * D;
    float*          c2  = (float*)(clo + (size_t)K * D);

    // Output layout: int64 bins[N] then f32 residual[N*D]
    long long* bins  = (long long*)d_out;
    float*     resid = (float*)((char*)d_out + (size_t)N * 8);

    {
        int threads = K * 32;                 // one wave per centroid row
        int blocks  = (threads + 255) / 256;
        vq_prep_centroids<<<blocks, 256, 0, stream>>>(cent, chi, clo, c2, K);
    }
    {
        int tiles  = N / 32;                  // one wave per 32-row tile
        int blocks = tiles / 8;               // 8 waves per block
        if (K == 2048) {
            vq_encode<2048><<<blocks, 256, 0, stream>>>(action, chi, clo, c2,
                                                        cent, bins, resid, K);
        } else {
            vq_encode<0><<<blocks, 256, 0, stream>>>(action, chi, clo, c2,
                                                     cent, bins, resid, K);
        }
    }
}
